// min_model_85830626443602
// MI455X (gfx1250) — compile-verified, matrix operand order checked by reading, not run on hardware
//
#include <hip/hip_runtime.h>

// Problem constants (fixed by the reference)
#define BB 8
#define TT 2048
#define VV 1024

typedef __bf16 bf16;
typedef __bf16  v16bf __attribute__((ext_vector_type(16)));
typedef float   v8f   __attribute__((ext_vector_type(8)));

// GEMM tiling: 128x128 block, 8 waves (4 M x 2 N), each wave 32x64 = 8 WMMA tiles
#define BM 128
#define BN 128
#define BK 32
#define SA (BK + 8)   // LDS A row stride: 40 bf16 = 80 B (16B-aligned rows)
#define SBT (BK + 8)  // LDS B^T row stride (B stored [n][k]): 40 bf16

// ---------------------------------------------------------------------------
// prep: bf16 conversions / transposes of the weights
//   WqT[q,o] = Wq[o,q],  Wkb[o,c] = Wk[o,c],  WvT[c,o] = Wv[o,c]
// ---------------------------------------------------------------------------
__global__ __launch_bounds__(256) void prep_kernel(
    const float* __restrict__ Wq, const float* __restrict__ Wk,
    const float* __restrict__ Wv, bf16* __restrict__ WqT,
    bf16* __restrict__ Wkb, bf16* __restrict__ WvT)
{
    int id = blockIdx.x * 256 + threadIdx.x;   // 0 .. V*V-1
    int r = id / VV;
    int c = id % VV;
    Wkb[id] = (bf16)Wk[id];
    WqT[id] = (bf16)Wq[c * VV + r];
    WvT[id] = (bf16)Wv[c * VV + r];
}

// ---------------------------------------------------------------------------
// scatter_L: L[b,t,c] = sum_{s<=t, idx[b,s]==c} v[t-s]   (bf16 out)
// One workgroup per (b,t) row; LDS f32 accumulation via ds_add_f32.
// ---------------------------------------------------------------------------
__global__ __launch_bounds__(256) void scatter_L_kernel(
    const int* __restrict__ idx, const float* __restrict__ v,
    bf16* __restrict__ L)
{
    __shared__ float row[VV];
    const int bt = blockIdx.x;
    const int b = bt / TT;
    const int t = bt % TT;
    for (int i = threadIdx.x; i < VV; i += 256) row[i] = 0.0f;
    __syncthreads();
    const int* ib = idx + b * TT;
    for (int s = threadIdx.x; s <= t; s += 256)
        atomicAdd(&row[ib[s]], v[t - s]);
    __syncthreads();
    bf16* out = L + ((size_t)b * TT + t) * VV;
    for (int i = threadIdx.x; i < VV; i += 256) out[i] = (bf16)row[i];
}

// ---------------------------------------------------------------------------
// Generic bf16 WMMA GEMM, C[128x128] block per workgroup (8 waves, wave32).
// B is staged into LDS transposed ([n][k]) so both A and B fragments are two
// contiguous 16B LDS loads per lane (ISA 16-bit layout: elems 0..7 <-> K=8h..,
// elems 8..15 <-> K=16+8h..).
// MODE 0: C_bf16[V,V]   = A[V,V] @ B[V,V]                       (M = WqT@Wk)
// MODE 1: att_bf16[T,T] = gatherA(M, idx_b) @ L_b^T, causal mask
// MODE 2: out_f32[T,V]  = att_b @ gatherB(WvT, idx_b) + 0.001, K clipped
// ---------------------------------------------------------------------------
template <int MODE>
__global__ __launch_bounds__(256) void gemm_kernel(
    const bf16* __restrict__ Abase, const bf16* __restrict__ Bbase,
    void* __restrict__ Cbase, const int* __restrict__ idx)
{
    __shared__ __align__(16) bf16 LA [BM * SA];
    __shared__ __align__(16) bf16 LBT[BN * SBT];

    const int tid = threadIdx.x;
    const int b  = blockIdx.z;
    const int m0 = blockIdx.y * BM;
    const int n0 = blockIdx.x * BN;

    const bf16* A;  const bf16* Bp;
    const int* amap = nullptr;  const int* bmap = nullptr;
    int lda, ldb, K;
    if (MODE == 0) { A = Abase; lda = VV; Bp = Bbase; ldb = VV; K = VV; }
    if (MODE == 1) { A = Abase; lda = VV; amap = idx + b * TT;
                     Bp = Bbase + (size_t)b * TT * VV; ldb = VV; K = VV; }
    if (MODE == 2) { A = Abase + (size_t)b * TT * TT; lda = TT;
                     Bp = Bbase; ldb = VV; bmap = idx + b * TT;
                     K = m0 + BM; }   // s > t contributes 0 (att masked)

    // Fully-masked attention block: just zero-fill (block-uniform branch).
    if (MODE == 1 && n0 > m0 + BM - 1) {
        bf16* C = (bf16*)Cbase + (size_t)b * TT * TT;
        uint4 z; z.x = 0; z.y = 0; z.z = 0; z.w = 0;
        #pragma unroll
        for (int p = 0; p < BM * BN / 8 / 256; p++) {
            int ch = tid + p * 256;
            int row = ch >> 4, seg = ch & 15;   // 16 chunks of 8 bf16 per row
            *(uint4*)(&C[(size_t)(m0 + row) * TT + n0 + seg * 8]) = z;
        }
        return;
    }

    const int lane = tid & 31;
    const int wave = tid >> 5;     // 0..7
    const int wm = wave >> 1;      // wave M row 0..3  (32 rows each)
    const int wn = wave & 1;       // wave N col 0..1  (64 cols each)
    const int h  = lane >> 4;      // K-half / M-half selector
    const int ln = lane & 15;

    v8f zero = {0, 0, 0, 0, 0, 0, 0, 0};
    v8f acc[2][4];
    #pragma unroll
    for (int i = 0; i < 2; i++)
        #pragma unroll
        for (int j = 0; j < 4; j++) acc[i][j] = zero;

    for (int k0 = 0; k0 < K; k0 += BK) {
        // ---- stage A block (BM x BK) into LDS [m][k], 16B vector copies ----
        #pragma unroll
        for (int p = 0; p < BM * BK / 8 / 256; p++) {  // 2 passes
            int seg = tid + p * 256;
            int row = seg >> 2;               // BK/8 = 4 segs per row
            int c8  = seg & 3;
            int grow = m0 + row;
            int arow = amap ? amap[grow] : grow;
            *(uint4*)(&LA[row * SA + c8 * 8]) =
                *(const uint4*)(A + (size_t)arow * lda + k0 + c8 * 8);
        }
        // ---- stage B block into LDS transposed [n][k] ----
        if (MODE == 1) {
            // Source L is [s=n][c=k] row-major already: pure vector copies.
            #pragma unroll
            for (int p = 0; p < BN * BK / 8 / 256; p++) {  // 2 passes
                int seg = tid + p * 256;
                int nrow = seg >> 2;          // 0..127
                int kseg = seg & 3;           // 0..3
                *(uint4*)(&LBT[nrow * SBT + kseg * 8]) =
                    *(const uint4*)(Bp + (size_t)(n0 + nrow) * ldb + k0 + kseg * 8);
            }
        } else {
            // Source B is [k][n] row-major: coalesced read, transposed write.
            #pragma unroll
            for (int p = 0; p < BK * BN / 8 / 256; p++) {  // 2 passes
                int seg = tid + p * 256;
                int krow = seg >> 4;          // 0..31 (16 segs per k row)
                int n8   = seg & 15;          // 0..15
                int gk   = k0 + krow;
                int brow = bmap ? bmap[gk] : gk;
                uint4 d = *(const uint4*)(Bp + (size_t)brow * ldb + n0 + n8 * 8);
                const bf16* dv = (const bf16*)&d;
                #pragma unroll
                for (int j = 0; j < 8; j++)
                    LBT[(n8 * 8 + j) * SBT + krow] = dv[j];
            }
        }
        __syncthreads();

        // ---- fragments: two contiguous 16B LDS loads each ----
        v16bf af[2], bfr[4];
        #pragma unroll
        for (int i = 0; i < 2; i++) {
            int mrow = wm * 32 + i * 16 + ln;
            uint4* ap = (uint4*)&af[i];
            ap[0] = *(const uint4*)(&LA[mrow * SA + 8 * h]);       // K = 8h..8h+7
            ap[1] = *(const uint4*)(&LA[mrow * SA + 16 + 8 * h]);  // K = 16+8h..23+8h
        }
        #pragma unroll
        for (int j = 0; j < 4; j++) {
            int ncol = wn * 64 + j * 16 + ln;
            uint4* bp = (uint4*)&bfr[j];
            bp[0] = *(const uint4*)(&LBT[ncol * SBT + 8 * h]);
            bp[1] = *(const uint4*)(&LBT[ncol * SBT + 16 + 8 * h]);
        }
        #pragma unroll
        for (int i = 0; i < 2; i++)
            #pragma unroll
            for (int j = 0; j < 4; j++)
                acc[i][j] = __builtin_amdgcn_wmma_f32_16x16x32_bf16(
                    false, af[i], false, bfr[j], (short)0, acc[i][j],
                    false, false);
        __syncthreads();
    }

    // ---- store D: lane layout m = r + 8h, n = lane%16 ----
    #pragma unroll
    for (int i = 0; i < 2; i++) {
        #pragma unroll
        for (int j = 0; j < 4; j++) {
            #pragma unroll
            for (int r = 0; r < 8; r++) {
                int mg = m0 + wm * 32 + i * 16 + r + 8 * h;
                int ng = n0 + wn * 64 + j * 16 + ln;
                float val = acc[i][j][r];
                if (MODE == 0) {
                    ((bf16*)Cbase)[(size_t)mg * VV + ng] = (bf16)val;
                }
                if (MODE == 1) {
                    bf16* C = (bf16*)Cbase + (size_t)b * TT * TT;
                    C[(size_t)mg * TT + ng] =
                        (ng <= mg) ? (bf16)val : (bf16)0.0f;
                }
                if (MODE == 2) {
                    float* C = (float*)Cbase + (size_t)b * TT * VV;
                    C[(size_t)mg * VV + ng] = val + 0.001f;
                }
            }
        }
    }
}

// ---------------------------------------------------------------------------
// Launch pipeline. Inputs (reference order): idx[B,T] i32, v[T,1] f32,
// Wk[V,V] f32, Wq[V,V] f32, Wv[V,V] f32. Output: logits [B,T,V] f32.
// Workspace layout (~104 MB): WqT | Wkb | WvT | M | L | att   (all bf16)
// ---------------------------------------------------------------------------
extern "C" void kernel_launch(void* const* d_in, const int* in_sizes, int n_in,
                              void* d_out, int out_size, void* d_ws, size_t ws_size,
                              hipStream_t stream)
{
    const int*   idx = (const int*)d_in[0];
    const float* v   = (const float*)d_in[1];
    const float* Wk  = (const float*)d_in[2];
    const float* Wq  = (const float*)d_in[3];
    const float* Wv  = (const float*)d_in[4];
    float* out = (float*)d_out;

    char* ws = (char*)d_ws;
    const size_t VVb = (size_t)VV * VV * sizeof(bf16);        // 2 MB
    bf16* WqT = (bf16*)ws;                 ws += VVb;
    bf16* Wkb = (bf16*)ws;                 ws += VVb;
    bf16* WvT = (bf16*)ws;                 ws += VVb;
    bf16* Mb  = (bf16*)ws;                 ws += VVb;
    bf16* Lb  = (bf16*)ws;                 ws += (size_t)BB * TT * VV * sizeof(bf16); // 33.5 MB
    bf16* att = (bf16*)ws;                 // B*T*T bf16 = 67 MB

    // 1) bf16 weight prep
    prep_kernel<<<(VV * VV) / 256, 256, 0, stream>>>(Wq, Wk, Wv, WqT, Wkb, WvT);
    // 2) M = WqT @ Wk   [V,V] bf16
    gemm_kernel<0><<<dim3(VV / BN, VV / BM, 1), 256, 0, stream>>>(WqT, Wkb, Mb, idx);
    // 3) L = layer_one (scatter-add), bf16 [B,T,V]
    scatter_L_kernel<<<BB * TT, 256, 0, stream>>>(idx, v, Lb);
    // 4) att[b] = gather_rows(M, idx_b) @ L[b]^T, causal-masked bf16 [B,T,T]
    gemm_kernel<1><<<dim3(TT / BN, TT / BM, BB), 256, 0, stream>>>(Mb, Lb, att, idx);
    // 5) out[b] = att[b] @ gather_rows(WvT, idx_b) + 0.001, f32 [B,T,V]
    gemm_kernel<2><<<dim3(VV / BN, TT / BM, BB), 256, 0, stream>>>(att, WvT, out, idx);
}
